// Attention_35605278883932
// MI455X (gfx1250) — matrix operand order repaired and checked by reading, not compiled, not run on hardware
//
#include <hip/hip_runtime.h>

// Problem constants (B, N, C, H, P from the reference)
#define B_  2
#define N_  2048
#define C_  256
#define H_  8
#define HD_ 32
#define P_  3
#define PH_ 32

typedef _Float16 h8  __attribute__((ext_vector_type(8)));
typedef _Float16 h16 __attribute__((ext_vector_type(16)));
typedef float    f8  __attribute__((ext_vector_type(8)));

static __device__ __forceinline__ h8 ld8(const _Float16* p) {
    return *(const h8*)p;
}
static __device__ __forceinline__ h16 cat16(h8 a, h8 b) {
    h16 r;
#pragma unroll
    for (int i = 0; i < 8; ++i) { r[i] = a[i]; r[i + 8] = b[i]; }
    return r;
}
static __device__ __forceinline__ f8 f8zero() {
    f8 z = {0.f, 0.f, 0.f, 0.f, 0.f, 0.f, 0.f, 0.f};
    return z;
}
// D = A(16x32 f16) x B(32x16 f16) + C(16x16 f32)
static __device__ __forceinline__ f8 wmma_f16(h16 a, h16 b, f8 c) {
    return __builtin_amdgcn_wmma_f32_16x16x32_f16(
        /*neg_a=*/false, a, /*neg_b=*/false, b,
        /*c_mod=*/(short)0, c, /*reuse_a=*/false, /*reuse_b=*/false);
}

// ---------------------------------------------------------------- conversions
__global__ void cvt_f32_to_f16(const float* __restrict__ src,
                               _Float16* __restrict__ dst, int n) {
    int t = blockIdx.x * blockDim.x + threadIdx.x;
    if (t < n) dst[t] = (_Float16)src[t];
}

// W stored row-major [K(in), N(out)]; produce WT f16 row-major [N][K]
__global__ void transpose_cvt(const float* __restrict__ W,
                              _Float16* __restrict__ WT, int K, int Ncol) {
    int t = blockIdx.x * blockDim.x + threadIdx.x;
    if (t < K * Ncol) {
        int k = t / Ncol, n = t % Ncol;
        WT[n * K + k] = (_Float16)W[t];
    }
}

// ------------------------------------------------------- tiny positional path
// ph[b,i,h] = ((relu(pos@W1+b1))@W2 + b2) @ Wh      (bh cancels in softmax)
__global__ void pos_ph_kernel(const float* __restrict__ pos,
                              const float* __restrict__ W1,
                              const float* __restrict__ b1,
                              const float* __restrict__ W2,
                              const float* __restrict__ b2,
                              const float* __restrict__ Wh,
                              float* __restrict__ ph) {
    int row = blockIdx.x * blockDim.x + threadIdx.x;
    if (row >= B_ * N_) return;
    float p0 = pos[row * 3 + 0], p1 = pos[row * 3 + 1], p2 = pos[row * 3 + 2];
    float t[3];
#pragma unroll
    for (int j = 0; j < 3; ++j) {
        float v = p0 * W1[0 * 3 + j] + p1 * W1[1 * 3 + j] + p2 * W1[2 * 3 + j] + b1[j];
        t[j] = v > 0.f ? v : 0.f;
    }
    float acc[H_];
#pragma unroll
    for (int h = 0; h < H_; ++h) acc[h] = 0.f;
    for (int m = 0; m < PH_; ++m) {
        float pm = t[0] * W2[0 * PH_ + m] + t[1] * W2[1 * PH_ + m] +
                   t[2] * W2[2 * PH_ + m] + b2[m];
#pragma unroll
        for (int h = 0; h < H_; ++h) acc[h] += pm * Wh[m * H_ + h];
    }
#pragma unroll
    for (int h = 0; h < H_; ++h) ph[row * H_ + h] = acc[h];
}

// ----------------------------------------------------------- qkv = x @ Ws
// Writes two f16 layouts: qkv_h [B,H,N,HD] and qkvT [B,H,HD,N]
__global__ void __launch_bounds__(128) gemm_qkv(
    const _Float16* __restrict__ xh, const _Float16* __restrict__ WsT,
    _Float16* __restrict__ qkv_h, _Float16* __restrict__ qkvT) {
    int wave = (blockIdx.x * blockDim.x + threadIdx.x) >> 5;
    int l = threadIdx.x & 31, l2 = l & 15, hi = l >> 4;
    int tN = wave & 15, tM = wave >> 4;     // 16 N-tiles, 256 M-tiles
    const _Float16* arow = xh + (size_t)(tM * 16 + l2) * C_;
    const _Float16* brow = WsT + (size_t)(tN * 16 + l2) * C_;
    f8 acc = f8zero();
#pragma unroll
    for (int kk = 0; kk < C_; kk += 32) {
        h16 a = cat16(ld8(arow + kk + hi * 8), ld8(arow + kk + 16 + hi * 8));
        h16 b = cat16(ld8(brow + kk + hi * 8), ld8(brow + kk + 16 + hi * 8));
        acc = wmma_f16(a, b, acc);
    }
#pragma unroll
    for (int r = 0; r < 8; ++r) {
        int gm = tM * 16 + r + hi * 8;      // global row = b*N + i
        int gc = tN * 16 + l2;              // h*HD + d
        int bb = gm >> 11, i = gm & (N_ - 1);
        int h = gc >> 5, d = gc & 31;
        _Float16 v = (_Float16)acc[r];
        qkv_h[(((size_t)(bb * H_ + h)) * N_ + i) * HD_ + d] = v;
        qkvT[(((size_t)(bb * H_ + h)) * HD_ + d) * N_ + i] = v;
    }
}

// -------------------------------------------- pv[b,h,:] = softmax(-ph) @ qkv
__global__ void pv_kernel(const float* __restrict__ ph,
                          const _Float16* __restrict__ qkv_h,
                          float* __restrict__ pv) {
    int bh = blockIdx.x;
    int b = bh / H_, h = bh % H_;
    int tid = threadIdx.x;
    __shared__ float red[256];
    __shared__ float s_acc[PH_ + 1];

    float lm = -1e30f;
    for (int j = tid; j < N_; j += 256)
        lm = fmaxf(lm, -ph[(b * N_ + j) * H_ + h]);
    red[tid] = lm;
    __syncthreads();
    for (int s = 128; s > 0; s >>= 1) {
        if (tid < s) red[tid] = fmaxf(red[tid], red[tid + s]);
        __syncthreads();
    }
    float gmax = red[0];
    if (tid < PH_ + 1) s_acc[tid] = 0.f;
    __syncthreads();

    float se = 0.f;
    float lpv[PH_];
#pragma unroll
    for (int d = 0; d < PH_; ++d) lpv[d] = 0.f;
    const _Float16* base = qkv_h + ((size_t)(b * H_ + h)) * N_ * HD_;
    for (int j = tid; j < N_; j += 256) {
        float w = __expf(-ph[(b * N_ + j) * H_ + h] - gmax);
        se += w;
        const h8* vp = (const h8*)(base + (size_t)j * HD_);
#pragma unroll
        for (int c = 0; c < 4; ++c) {
            h8 v = vp[c];
#pragma unroll
            for (int u = 0; u < 8; ++u) lpv[c * 8 + u] += w * (float)v[u];
        }
    }
    atomicAdd(&s_acc[PH_], se);
#pragma unroll
    for (int d = 0; d < PH_; ++d) atomicAdd(&s_acc[d], lpv[d]);
    __syncthreads();
    if (tid < PH_) pv[(b * H_ + h) * HD_ + tid] = s_acc[tid] / s_acc[PH_];
}

// ----------------------------------------------------- fused flash attention
// One wave handles 16 query rows; block = 4 waves = 64 rows of one (b,h).
// comb[b,i,h*HD+d] = (1-g_h)*softmax(QK^T/sqrt(hd))V + g_h*pv[b,h,d]   (f16)
__global__ void __launch_bounds__(128) flash_kernel(
    const _Float16* __restrict__ qkv_h, const _Float16* __restrict__ qkvT,
    const float* __restrict__ pv, const float* __restrict__ gate,
    _Float16* __restrict__ comb) {
    __shared__ _Float16 lds_p[4 * 16 * 40];   // per-wave 16x32 P tile, stride 40
    const float SCALE = 0.17677669529663687f; // 1/sqrt(32)

    int w = threadIdx.x >> 5;
    int l = threadIdx.x & 31, l2 = l & 15, hi = l >> 4;
    int bid = blockIdx.x;
    int qb = bid & 31;
    int h = (bid >> 5) & 7;
    int bb = bid >> 8;
    int i0 = qb * 64 + w * 16;

    const _Float16* headQ  = qkv_h + ((size_t)(bb * H_ + h)) * N_ * HD_;
    const _Float16* headVT = qkvT + ((size_t)(bb * H_ + h)) * HD_ * N_;

    // A-fragment of Q (rows i0..i0+15), loaded once
    const _Float16* qp = headQ + (size_t)(i0 + l2) * HD_;
    h16 qfrag = cat16(ld8(qp + hi * 8), ld8(qp + 16 + hi * 8));

    f8 o0 = f8zero(), o1 = f8zero();
    float mrow[8], lrow[8];
#pragma unroll
    for (int r = 0; r < 8; ++r) { mrow[r] = -1e30f; lrow[r] = 0.f; }
    _Float16* lp = lds_p + w * 16 * 40;

    for (int j0 = 0; j0 < N_; j0 += 32) {
        // B-fragments of K^T (keys j0..j0+31): lane = key column, hd in table order
        const _Float16* kp = headQ + (size_t)j0 * HD_;
        h16 kb0 = cat16(ld8(kp + l2 * HD_ + hi * 8),
                        ld8(kp + l2 * HD_ + 16 + hi * 8));
        h16 kb1 = cat16(ld8(kp + (16 + l2) * HD_ + hi * 8),
                        ld8(kp + (16 + l2) * HD_ + 16 + hi * 8));
        f8 s0 = wmma_f16(qfrag, kb0, f8zero());
        f8 s1 = wmma_f16(qfrag, kb1, f8zero());

        float mt[8];
#pragma unroll
        for (int r = 0; r < 8; ++r) {
            s0[r] *= SCALE; s1[r] *= SCALE;
            mt[r] = fmaxf(s0[r], s1[r]);
        }
#pragma unroll
        for (int off = 1; off < 16; off <<= 1)
#pragma unroll
            for (int r = 0; r < 8; ++r)
                mt[r] = fmaxf(mt[r], __shfl_xor(mt[r], off, 32));

        float al[8], st[8];
#pragma unroll
        for (int r = 0; r < 8; ++r) {
            float mn = fmaxf(mrow[r], mt[r]);
            al[r] = __expf(mrow[r] - mn);
            mrow[r] = mn;
            s0[r] = __expf(s0[r] - mn);
            s1[r] = __expf(s1[r] - mn);
            st[r] = s0[r] + s1[r];
        }
#pragma unroll
        for (int off = 1; off < 16; off <<= 1)
#pragma unroll
            for (int r = 0; r < 8; ++r)
                st[r] += __shfl_xor(st[r], off, 32);
#pragma unroll
        for (int r = 0; r < 8; ++r) {
            lrow[r] = lrow[r] * al[r] + st[r];
            o0[r] *= al[r];
            o1[r] *= al[r];
        }

        // Stage P (C-layout) -> LDS -> reload as A-fragment (wave-private, no barrier)
#pragma unroll
        for (int r = 0; r < 8; ++r) {
            int q = r + hi * 8;
            lp[q * 40 + l2]      = (_Float16)s0[r];
            lp[q * 40 + 16 + l2] = (_Float16)s1[r];
        }
        h16 pa = cat16(*(const h8*)(lp + l2 * 40 + hi * 8),
                       *(const h8*)(lp + l2 * 40 + 16 + hi * 8));

        // B-fragments of V (keys j0..j0+31) from transposed copy: keys in table order
        const _Float16* v0p = headVT + (size_t)l2 * N_ + j0 + hi * 8;
        const _Float16* v1p = headVT + (size_t)(16 + l2) * N_ + j0 + hi * 8;
        h16 vb0 = cat16(ld8(v0p), ld8(v0p + 16));
        h16 vb1 = cat16(ld8(v1p), ld8(v1p + 16));
        o0 = wmma_f16(pa, vb0, o0);
        o1 = wmma_f16(pa, vb1, o1);
    }

    // Epilogue: normalize, gate, add positional rank-1 term, emit f16
    float g = 1.f / (1.f + __expf(-gate[h]));
    float pv0 = pv[(bb * H_ + h) * HD_ + l2];
    float pv1 = pv[(bb * H_ + h) * HD_ + 16 + l2];
#pragma unroll
    for (int r = 0; r < 8; ++r) {
        int i = i0 + r + hi * 8;
        float inv = 1.f / lrow[r];
        size_t base = ((size_t)(bb * N_ + i)) * C_ + h * HD_;
        comb[base + l2]      = (_Float16)((1.f - g) * o0[r] * inv + g * pv0);
        comb[base + 16 + l2] = (_Float16)((1.f - g) * o1[r] * inv + g * pv1);
    }
}

// ------------------------------------------------------ out = comb @ Wo + bo
__global__ void __launch_bounds__(128) gemm_out(
    const _Float16* __restrict__ comb, const _Float16* __restrict__ WoT,
    const float* __restrict__ bo, float* __restrict__ out) {
    int wave = (blockIdx.x * blockDim.x + threadIdx.x) >> 5;
    int l = threadIdx.x & 31, l2 = l & 15, hi = l >> 4;
    int tN = wave & 15, tM = wave >> 4;
    const _Float16* arow = comb + (size_t)(tM * 16 + l2) * C_;
    const _Float16* brow = WoT + (size_t)(tN * 16 + l2) * C_;
    f8 acc = f8zero();
#pragma unroll
    for (int kk = 0; kk < C_; kk += 32) {
        h16 a = cat16(ld8(arow + kk + hi * 8), ld8(arow + kk + 16 + hi * 8));
        h16 b = cat16(ld8(brow + kk + hi * 8), ld8(brow + kk + 16 + hi * 8));
        acc = wmma_f16(a, b, acc);
    }
#pragma unroll
    for (int r = 0; r < 8; ++r) {
        int gm = tM * 16 + r + hi * 8;
        int gc = tN * 16 + l2;
        out[(size_t)gm * C_ + gc] = acc[r] + bo[gc];
    }
}

// ---------------------------------------------------------------- launcher
extern "C" void kernel_launch(void* const* d_in, const int* in_sizes, int n_in,
                              void* d_out, int out_size, void* d_ws, size_t ws_size,
                              hipStream_t stream) {
    (void)in_sizes; (void)n_in; (void)out_size; (void)ws_size;
    const float* x    = (const float*)d_in[0];
    const float* pos  = (const float*)d_in[1];
    const float* Ws   = (const float*)d_in[2];
    const float* W1   = (const float*)d_in[3];
    const float* b1   = (const float*)d_in[4];
    const float* W2   = (const float*)d_in[5];
    const float* b2   = (const float*)d_in[6];
    const float* Wh   = (const float*)d_in[7];
    // d_in[8] (bh) cancels inside the positional softmax
    const float* gate = (const float*)d_in[9];
    const float* Wo   = (const float*)d_in[10];
    const float* bo   = (const float*)d_in[11];
    float* out = (float*)d_out;

    char* ws = (char*)d_ws;
    size_t off = 0;
    auto take = [&](size_t bytes) -> char* {
        char* p = ws + off;
        off = (off + bytes + 255) & ~(size_t)255;
        return p;
    };
    _Float16* xh   = (_Float16*)take((size_t)B_ * N_ * C_ * 2);
    _Float16* WsT  = (_Float16*)take((size_t)C_ * C_ * 2);
    _Float16* WoT  = (_Float16*)take((size_t)C_ * C_ * 2);
    float*    ph   = (float*)   take((size_t)B_ * N_ * H_ * 4);
    _Float16* qkvh = (_Float16*)take((size_t)B_ * H_ * N_ * HD_ * 2);
    _Float16* qkvT = (_Float16*)take((size_t)B_ * H_ * N_ * HD_ * 2);
    float*    pvb  = (float*)   take((size_t)B_ * H_ * HD_ * 4);
    _Float16* comb = (_Float16*)take((size_t)B_ * N_ * C_ * 2);

    cvt_f32_to_f16<<<(B_ * N_ * C_ + 255) / 256, 256, 0, stream>>>(x, xh, B_ * N_ * C_);
    transpose_cvt<<<(C_ * C_ + 255) / 256, 256, 0, stream>>>(Ws, WsT, C_, C_);
    transpose_cvt<<<(C_ * C_ + 255) / 256, 256, 0, stream>>>(Wo, WoT, C_, C_);
    pos_ph_kernel<<<(B_ * N_ + 255) / 256, 256, 0, stream>>>(pos, W1, b1, W2, b2, Wh, ph);

    int gemm_blocks = (B_ * N_ / 16) * (C_ / 16) / 4;  // 4 waves/block
    gemm_qkv<<<gemm_blocks, 128, 0, stream>>>(xh, WsT, qkvh, qkvT);
    pv_kernel<<<B_ * H_, 256, 0, stream>>>(ph, qkvh, pvb);
    flash_kernel<<<B_ * H_ * (N_ / 64), 128, 0, stream>>>(qkvh, qkvT, pvb, gate, comb);
    gemm_out<<<gemm_blocks, 128, 0, stream>>>(comb, WoT, bo, out);
}
